// FeatureAttentionLayer_16947940950502
// MI455X (gfx1250) — matrix operand order repaired and checked by reading, not compiled, not run on hardware
//
#include <hip/hip_runtime.h>

typedef __attribute__((ext_vector_type(16))) _Float16 v16h;
typedef __attribute__((ext_vector_type(8)))  _Float16 v8h;
typedef __attribute__((ext_vector_type(8)))  float    v8f;
typedef __attribute__((ext_vector_type(4)))  float    v4f;

#define ALPHA 0.2f
#define BB 128
#define WW 128
#define NN 512
#define PITCH 520   // f16 pitch for xpT rows: 1040 B/row -> 16B aligned, 4-bank lane stride

// ---------------------------------------------------------------------------
// Kernel 1: c[w] = sum_v W_fc[v,w]*a2[v];  d = b_fc . a2
// ---------------------------------------------------------------------------
__global__ void prep_cd(const float* __restrict__ W_fc,
                        const float* __restrict__ b_fc,
                        const float* __restrict__ attn,
                        float* __restrict__ ws) {
    int w = threadIdx.x;  // 128 threads
    float acc = 0.f;
    #pragma unroll 4
    for (int v = 0; v < WW; ++v) acc += W_fc[v * WW + w] * attn[WW + v];
    ws[w] = acc;
    if (w == 0) {
        float d = 0.f;
        for (int v = 0; v < WW; ++v) d += b_fc[v] * attn[WW + v];
        ws[WW] = d;
    }
}

// ---------------------------------------------------------------------------
// Kernel 2: S1[b,n] = x[b,:,n].a1 ; S2[b,n] = x[b,:,n].c + d
// ---------------------------------------------------------------------------
__global__ void compute_s12(const float* __restrict__ x,
                            const float* __restrict__ attn,
                            const float* __restrict__ ws,
                            float* __restrict__ S1,
                            float* __restrict__ S2) {
    __shared__ float a1l[WW];
    __shared__ float cl[WW];
    int t = threadIdx.x;                  // 256 threads
    if (t < WW) a1l[t] = attn[t];
    else        cl[t - WW] = ws[t - WW];
    __syncthreads();

    int gid = blockIdx.x * 256 + t;       // b*512 + n
    int b = gid >> 9;
    int n = gid & 511;
    const float* xc = x + (size_t)b * (WW * NN) + n;
    float s1 = 0.f, s2 = 0.f;
    #pragma unroll 4
    for (int w = 0; w < WW; ++w) {
        float xv = xc[(size_t)w * NN];    // coalesced across n
        s1 += xv * a1l[w];
        s2 += xv * cl[w];
    }
    S1[gid] = s1;
    S2[gid] = s2 + ws[WW];
}

// ---------------------------------------------------------------------------
// Kernel 3: fused softmax( lrelu(s1_i + s2_j) ) @ xp  -> sigmoid -> transpose
// One block = one batch x 128 i-rows. 8 waves, each wave owns a 16-row M tile.
// ---------------------------------------------------------------------------
__global__ void __launch_bounds__(256)
attn_gemm(const float* __restrict__ x,
          const float* __restrict__ S1,
          const float* __restrict__ S2,
          float* __restrict__ out) {
    extern __shared__ char smem[];
    _Float16* xpT = (_Float16*)smem;                       // [128][PITCH] f16
    float* s2l  = (float*)(smem + (size_t)WW * PITCH * 2); // [512]
    float* redl = s2l + NN;                                // [8]

    const int tid  = threadIdx.x;
    const int lane = tid & 31;
    const int wave = tid >> 5;
    const int hlf  = lane >> 4;      // 0: lanes 0-15, 1: lanes 16-31
    const int ln16 = lane & 15;

    const int b     = blockIdx.x >> 2;
    const int itile = blockIdx.x & 3;
    const int i0    = itile * 128 + wave * 16;   // this wave's 16 output rows

    // ---- stage x[b] (== xp^T, [w][j]) into LDS as f16 ----
    const float* xb = x + (size_t)b * (WW * NN);
    for (int k = tid; k < (WW * NN) / 4; k += 256) {
        int e = k * 4;
        int r = e >> 9;               // w row
        int c = e & 511;              // j col
        v4f xv = *(const v4f*)(xb + e);
        _Float16* dst = xpT + r * PITCH + c;
        dst[0] = (_Float16)xv.x; dst[1] = (_Float16)xv.y;
        dst[2] = (_Float16)xv.z; dst[3] = (_Float16)xv.w;
    }
    // ---- stage s2 row for this batch ----
    const float* S2b = S2 + b * NN;
    for (int k = tid; k < NN; k += 256) s2l[k] = S2b[k];
    __syncthreads();

    // ---- M2 = max_j s2  (lrelu monotonic => rowmax_i = lrelu(s1_i + M2)) ----
    float m = fmaxf(s2l[2 * tid], s2l[2 * tid + 1]);
    #pragma unroll
    for (int off = 16; off > 0; off >>= 1) m = fmaxf(m, __shfl_xor(m, off, 32));
    if (lane == 0) redl[wave] = m;
    __syncthreads();
    float M2 = redl[0];
    #pragma unroll
    for (int wv = 1; wv < 8; ++wv) M2 = fmaxf(M2, redl[wv]);

    // per-lane row scalars (lane -> row ln16 of this wave's M tile)
    const float s1v  = S1[b * NN + i0 + ln16];
    const float tmax = s1v + M2;
    const float mrow = tmax > 0.f ? tmax : ALPHA * tmax;

    v8f acc[8];
    #pragma unroll
    for (int n = 0; n < 8; ++n)
        #pragma unroll
        for (int e = 0; e < 8; ++e) acc[n][e] = 0.f;

    float psum = 0.f;   // partial softmax denominator (this lane's K subset)

    // ---- K loop: 16 chunks of 32 j's ----
    for (int chunk = 0; chunk < 16; ++chunk) {
        const int j0 = chunk * 32;
        // Build A fragment (16x32 f16) in ISA lane layout:
        //   lanes 0-15 : K in {0..7, 16..23};  lanes 16-31 : K in {8..15, 24..31}
        v16h af;
        #pragma unroll
        for (int e = 0; e < 16; ++e) {
            int K = (e < 8) ? (hlf * 8 + e) : (16 + hlf * 8 + (e - 8));
            float t = s1v + s2l[j0 + K];
            t = t > 0.f ? t : ALPHA * t;
            float p = __expf(t - mrow);
            _Float16 ph = (_Float16)p;
            psum += (float)ph;        // denominator matches f16 numerator
            af[e] = ph;
        }
        // 8 N-tiles of 16 columns each
        #pragma unroll
        for (int n = 0; n < 8; ++n) {
            // B fragment (32x16): lane = column w, elems = K {8h..8h+7, 16+8h..16+8h+7}
            const _Float16* bp = xpT + (n * 16 + ln16) * PITCH + j0 + hlf * 8;
            v8h blo = *(const v8h*)bp;
            v8h bhi = *(const v8h*)(bp + 16);
            v16h bf = __builtin_shufflevector(blo, bhi,
                0, 1, 2, 3, 4, 5, 6, 7, 8, 9, 10, 11, 12, 13, 14, 15);
            acc[n] = __builtin_amdgcn_wmma_f32_16x16x32_f16(
                false, af, false, bf, (short)0, acc[n], false, false);
        }
    }

    // ---- softmax denominator: combine the two K-half lanes of each row ----
    float fullsum = psum + __shfl_xor(psum, 16, 32);
    float rs[8];
    #pragma unroll
    for (int r = 0; r < 8; ++r)
        rs[r] = __shfl(fullsum, r + 8 * hlf, 32);  // rowsum for D row r+8*hlf

    // ---- epilogue: normalize, sigmoid, store transposed (B, W, N) ----
    #pragma unroll
    for (int n = 0; n < 8; ++n) {
        int w = n * 16 + ln16;                     // D column = lane
        float* op = out + ((size_t)b * WW + w) * NN + i0 + 8 * hlf;
        #pragma unroll
        for (int r = 0; r < 8; ++r) {              // D VGPR r = row r+8*hlf
            float v = acc[n][r] / rs[r];
            v = 1.f / (1.f + __expf(-v));
            op[r] = v;
        }
    }
}

// ---------------------------------------------------------------------------
extern "C" void kernel_launch(void* const* d_in, const int* in_sizes, int n_in,
                              void* d_out, int out_size, void* d_ws, size_t ws_size,
                              hipStream_t stream) {
    const float* x    = (const float*)d_in[0];   // (128,128,512)
    const float* W_fc = (const float*)d_in[1];   // (128,128)
    const float* b_fc = (const float*)d_in[2];   // (128,)
    const float* attn = (const float*)d_in[3];   // (256,1)
    float* out = (float*)d_out;                  // (128,128,512)
    float* ws  = (float*)d_ws;

    float* S1 = ws + 256;                        // [B*N]
    float* S2 = S1 + BB * NN;                    // [B*N]

    prep_cd<<<1, 128, 0, stream>>>(W_fc, b_fc, attn, ws);
    compute_s12<<<(BB * NN) / 256, 256, 0, stream>>>(x, attn, ws, S1, S2);

    size_t smem = (size_t)WW * PITCH * 2 + (size_t)NN * 4 + 32;
    attn_gemm<<<BB * 4, 256, smem, stream>>>(x, S1, S2, out);
}
